// FP8Linear_40089224741095
// MI455X (gfx1250) — 8-bit matrix operand order repaired and checked by reading, compiled, not run on hardware
//
#include <hip/hip_runtime.h>

typedef __attribute__((ext_vector_type(16))) int   v16i;
typedef __attribute__((ext_vector_type(8)))  float v8f;

__device__ __forceinline__ unsigned f32_to_e4m3(float f) {
    unsigned u = __float_as_uint(f);
    unsigned s = (u >> 31) << 7;
    float a = fabsf(f);
    if (!(a < 448.0f)) return s | 0x7Eu;
    if (a < 0.015625f) {
        unsigned m = (unsigned)(int)rintf(a * 512.0f);
        return s | m;
    }
    unsigned au = u & 0x7FFFFFFFu;
    unsigned lsb = (au >> 20) & 1u;
    au += 0x7FFFFu + lsb;
    int e = (int)(au >> 23) - 127;
    if (e > 8) return s | 0x7Eu;
    unsigned m = (au >> 20) & 7u;
    return s | ((unsigned)(e + 7) << 3) | m;
}

__device__ __forceinline__ unsigned pack4_e4m3(float4 v, float inv) {
    return  f32_to_e4m3(v.x * inv)
         | (f32_to_e4m3(v.y * inv) << 8)
         | (f32_to_e4m3(v.z * inv) << 16)
         | (f32_to_e4m3(v.w * inv) << 24);
}

__global__ void fp8lin_init_kernel(unsigned* absmax_bits) {
    *absmax_bits = 0u;
}

__global__ __launch_bounds__(256)
void fp8lin_absmax_kernel(const float4* __restrict__ x, int n4,
                          unsigned* __restrict__ absmax_bits) {
    __shared__ float red[256];
    float m = 0.0f;
    for (int i = blockIdx.x * blockDim.x + threadIdx.x; i < n4;
         i += gridDim.x * blockDim.x) {
        float4 v = x[i];
        m = fmaxf(m, fmaxf(fmaxf(fabsf(v.x), fabsf(v.y)),
                           fmaxf(fabsf(v.z), fabsf(v.w))));
    }
    red[threadIdx.x] = m;
    __syncthreads();
    for (int sft = 128; sft > 0; sft >>= 1) {
        if ((int)threadIdx.x < sft)
            red[threadIdx.x] = fmaxf(red[threadIdx.x], red[threadIdx.x + sft]);
        __syncthreads();
    }
    if (threadIdx.x == 0)
        atomicMax(absmax_bits, __float_as_uint(red[0]));
}

__global__ __launch_bounds__(256)
void fp8lin_quantA_kernel(const float* __restrict__ x, unsigned* __restrict__ q,
                          int n_dw, int K,
                          const unsigned* __restrict__ absmax_bits) {
    int t = blockIdx.x * blockDim.x + threadIdx.x;
    if (t >= n_dw) return;
    float scale = fmaxf(__uint_as_float(*absmax_bits) * (1.0f / 448.0f), 1e-7f);
    float inv = 1.0f / scale;

    const int KB     = K >> 7;
    const int tile   = t >> 9;
    const int within = t & 511;
    const int lane   = within >> 4;
    const int v      = within & 15;
    const int mt     = tile / KB;
    const int kb     = tile - mt * KB;
    const int k = kb * 128 + ((v >> 3) << 6) + ((lane >> 4) << 3)
                + (((v & 7) >> 1) << 4) + ((v & 1) << 2);
    const int row = (mt << 4) + (lane & 15);
    float4 src = *reinterpret_cast<const float4*>(x + (size_t)row * (size_t)K + k);
    q[t] = pack4_e4m3(src, inv);
}

__global__ __launch_bounds__(256)
void fp8lin_quantB_kernel(const float* __restrict__ w, unsigned* __restrict__ q,
                          int n_dw, int K) {
    int t = blockIdx.x * blockDim.x + threadIdx.x;
    if (t >= n_dw) return;
    const int KB     = K >> 7;
    const int tile   = t >> 9;
    const int within = t & 511;
    const int lane   = within >> 4;
    const int v      = within & 15;
    const int nt     = tile / KB;
    const int kb     = tile - nt * KB;
    const int k = kb * 128 + ((v >> 3) << 6) + ((lane >> 4) << 3)
                + (((v & 7) >> 1) << 4) + ((v & 1) << 2);
    const int n = (nt << 4) + (lane & 15);
    float4 src = *reinterpret_cast<const float4*>(w + (size_t)n * (size_t)K + k);
    q[t] = pack4_e4m3(src, 1.0f);
}

__global__ __launch_bounds__(256)
void fp8lin_gemm_kernel(const unsigned char* __restrict__ xq,
                        const unsigned char* __restrict__ wq,
                        const float* __restrict__ wscale_p,
                        const float* __restrict__ bias,
                        float* __restrict__ out,
                        int M, int N, int K) {
    const int lane    = threadIdx.x & 31;
    const int wave    = threadIdx.x >> 5;
    const int laneRow = lane & 15;
    const int laneHi  = lane >> 4;

    const int wm = wave & 3;
    const int wn = wave >> 2;
    const int m_base = blockIdx.y * 128 + wm * 32;
    const int n_base = blockIdx.x * 128 + wn * 64;

    const int KB = K >> 7;
    const size_t tileStride = 2048;
    const size_t rowTileStr = (size_t)KB * 2048;

    const unsigned char* aPtr = xq + (size_t)(m_base >> 4) * rowTileStr + (size_t)lane * 64;
    const unsigned char* bPtr = wq + (size_t)(n_base >> 4) * rowTileStr + (size_t)lane * 64;

    v8f acc00 = {}, acc01 = {}, acc02 = {}, acc03 = {};
    v8f acc10 = {}, acc11 = {}, acc12 = {}, acc13 = {};

    for (int kb = 0; kb < KB; ++kb) {
        v16i a0, a1, b0, b1, b2, b3;
        const unsigned char* ap = aPtr + (size_t)kb * tileStride;
        const unsigned char* bp = bPtr + (size_t)kb * tileStride;
        #pragma unroll
        for (int c = 0; c < 4; ++c) {
            int4 ta0 = *reinterpret_cast<const int4*>(ap + c * 16);
            int4 ta1 = *reinterpret_cast<const int4*>(ap + rowTileStr + c * 16);
            a0[4 * c] = ta0.x; a0[4 * c + 1] = ta0.y; a0[4 * c + 2] = ta0.z; a0[4 * c + 3] = ta0.w;
            a1[4 * c] = ta1.x; a1[4 * c + 1] = ta1.y; a1[4 * c + 2] = ta1.z; a1[4 * c + 3] = ta1.w;
            int4 tb0 = *reinterpret_cast<const int4*>(bp + c * 16);
            int4 tb1 = *reinterpret_cast<const int4*>(bp + rowTileStr + c * 16);
            int4 tb2 = *reinterpret_cast<const int4*>(bp + 2 * rowTileStr + c * 16);
            int4 tb3 = *reinterpret_cast<const int4*>(bp + 3 * rowTileStr + c * 16);
            b0[4 * c] = tb0.x; b0[4 * c + 1] = tb0.y; b0[4 * c + 2] = tb0.z; b0[4 * c + 3] = tb0.w;
            b1[4 * c] = tb1.x; b1[4 * c + 1] = tb1.y; b1[4 * c + 2] = tb1.z; b1[4 * c + 3] = tb1.w;
            b2[4 * c] = tb2.x; b2[4 * c + 1] = tb2.y; b2[4 * c + 2] = tb2.z; b2[4 * c + 3] = tb2.w;
            b3[4 * c] = tb3.x; b3[4 * c + 1] = tb3.y; b3[4 * c + 2] = tb3.z; b3[4 * c + 3] = tb3.w;
        }
        acc00 = __builtin_amdgcn_wmma_f32_16x16x128_fp8_fp8(a0, b0, (short)0, acc00, false, false);
        acc01 = __builtin_amdgcn_wmma_f32_16x16x128_fp8_fp8(a0, b1, (short)0, acc01, false, false);
        acc02 = __builtin_amdgcn_wmma_f32_16x16x128_fp8_fp8(a0, b2, (short)0, acc02, false, false);
        acc03 = __builtin_amdgcn_wmma_f32_16x16x128_fp8_fp8(a0, b3, (short)0, acc03, false, false);
        acc10 = __builtin_amdgcn_wmma_f32_16x16x128_fp8_fp8(a1, b0, (short)0, acc10, false, false);
        acc11 = __builtin_amdgcn_wmma_f32_16x16x128_fp8_fp8(a1, b1, (short)0, acc11, false, false);
        acc12 = __builtin_amdgcn_wmma_f32_16x16x128_fp8_fp8(a1, b2, (short)0, acc12, false, false);
        acc13 = __builtin_amdgcn_wmma_f32_16x16x128_fp8_fp8(a1, b3, (short)0, acc13, false, false);
    }

    const float wscale = wscale_p[0];
    auto store_tile = [&](const v8f& acc, int mi, int ni) {
        const int col  = n_base + ni * 16 + laneRow;
        const float bv = bias[col];
        const int row0 = m_base + mi * 16 + laneHi * 8;
        #pragma unroll
        for (int r = 0; r < 8; ++r) {
            out[(size_t)(row0 + r) * (size_t)N + col] = acc[r] * wscale + bv;
        }
    };
    store_tile(acc00, 0, 0); store_tile(acc01, 0, 1);
    store_tile(acc02, 0, 2); store_tile(acc03, 0, 3);
    store_tile(acc10, 1, 0); store_tile(acc11, 1, 1);
    store_tile(acc12, 1, 2); store_tile(acc13, 1, 3);
}

extern "C" void kernel_launch(void* const* d_in, const int* in_sizes, int n_in,
                              void* d_out, int out_size, void* d_ws, size_t ws_size,
                              hipStream_t stream) {
    const float* x      = (const float*)d_in[0];
    const float* w      = (const float*)d_in[1];
    const float* wscale = (const float*)d_in[2];
    const float* bias   = (const float*)d_in[3];
    float*       out    = (float*)d_out;

    const long x_elems = (long)in_sizes[0];
    const long w_elems = (long)in_sizes[1];
    const int  N       = in_sizes[3];
    const int  K       = (int)(w_elems / (long)N);
    const int  M       = (int)(x_elems / (long)K);

    unsigned*      absmax_bits = (unsigned*)d_ws;
    unsigned char* xq = (unsigned char*)d_ws + 256;
    unsigned char* wq = xq + x_elems;

    fp8lin_init_kernel<<<1, 1, 0, stream>>>(absmax_bits);

    const int n4x = (int)(x_elems / 4);
    fp8lin_absmax_kernel<<<1024, 256, 0, stream>>>((const float4*)x, n4x, absmax_bits);

    fp8lin_quantA_kernel<<<(n4x + 255) / 256, 256, 0, stream>>>(
        x, (unsigned*)xq, n4x, K, absmax_bits);

    const int n4w = (int)(w_elems / 4);
    fp8lin_quantB_kernel<<<(n4w + 255) / 256, 256, 0, stream>>>(
        w, (unsigned*)wq, n4w, K);

    dim3 grid(N / 128, M / 128);
    fp8lin_gemm_kernel<<<grid, 256, 0, stream>>>(xq, wq, wscale, bias, out, M, N, K);
}
